// Seq2SeqAttentionSharedEmbeddingHidden_15023795602103
// MI455X (gfx1250) — compile-verified, matrix operand order checked by reading, not run on hardware
//
#include <hip/hip_runtime.h>

// ---------------------------------------------------------------------------
// CDNA5 (gfx1250) seq2seq-with-attention forward.
//   * all dense matmuls -> v_wmma_f32_16x16x32_bf16 (fp32 accumulate)
//   * fp32->bf16 conversions hoisted out of all inner loops (weights converted
//     once per launch; activations consumed as WMMA A-operands stored bf16)
//   * GEMM: 2-stage software pipeline (ping-pong fragment regs) + 16x64 wave
//     tile (A-fragment reused across 4 WMMAs) => loads overlap WMMA issue
//   * input-gate GEMMs hoisted out of the recurrences (batched over time)
//   * recurrences: persistent workgroups, hidden state LDS-resident bf16
//     (ds_load_b128 fragments), cell state in VGPRs across all 64 steps
//   * vocab GEMM (134 GFLOP) dominant; bf16 Wv (65MB) is L2-resident
// ---------------------------------------------------------------------------

typedef __attribute__((ext_vector_type(16))) __bf16 v16bf;
typedef __attribute__((ext_vector_type(8)))  float  v8f;

#define WS_B 32
#define WS_S 64
#define WS_EMB 512
#define WS_HID 1024
#define WS_SH  512    // per-direction encoder hidden
#define WS_V   32000

__device__ __forceinline__ float sigmoidf_(float x) {
  return 1.0f / (1.0f + __expf(-x));
}
__device__ __forceinline__ float tanhf_(float x) {
  x = fminf(15.0f, fmaxf(-15.0f, x));
  float e = __expf(2.0f * x);
  return (e - 1.0f) / (e + 1.0f);
}
// fp32 -> bf16 (round to nearest even); only used OUTSIDE inner loops
__device__ __forceinline__ unsigned short f2bfu(float f) {
  unsigned int u = __float_as_uint(f);
  unsigned int r = u + 0x7FFFu + ((u >> 16) & 1u);
  return (unsigned short)(r >> 16);
}
__device__ __forceinline__ float bfu2f(unsigned short s) {
  return __uint_as_float(((unsigned int)s) << 16);
}
// unpack a uint holding two packed bf16
__device__ __forceinline__ float bflo(unsigned int u) {
  return __uint_as_float(u << 16);
}
__device__ __forceinline__ float bfhi(unsigned int u) {
  return __uint_as_float(u & 0xffff0000u);
}

union BF16x16 {
  uint4 u[2];
  v16bf v;
};

// load 16 contiguous bf16 (32B, 16B-aligned) as a WMMA fragment; works for
// both global (global_load_b128 x2) and LDS (ds_load_b128 x2) pointers
__device__ __forceinline__ v16bf load16bf(const unsigned short* p) {
  BF16x16 r;
  r.u[0] = *(const uint4*)(p);
  r.u[1] = *(const uint4*)(p + 8);
  return r.v;
}
__device__ __forceinline__ v8f wmma_bf16(v16bf a, v16bf b, v8f c) {
  return __builtin_amdgcn_wmma_f32_16x16x32_bf16(false, a, false, b, (short)0, c,
                                                 false, false);
}
__device__ __forceinline__ v8f vzero8() {
  v8f z = {0.f, 0.f, 0.f, 0.f, 0.f, 0.f, 0.f, 0.f};
  return z;
}

// ---------------------------------------------------------------------------
// Streaming fp32 -> bf16 conversion (weights; once per launch)
// ---------------------------------------------------------------------------
__global__ void cvt_bf16_kernel(const float* __restrict__ in,
                                unsigned short* __restrict__ out, int n) {
  int i = blockIdx.x * blockDim.x + threadIdx.x;
  int stride = gridDim.x * blockDim.x;
  for (; i < n; i += stride) out[i] = f2bfu(in[i]);
}

// ---------------------------------------------------------------------------
// Embedding gather -> bf16, time-major: out[(s*B+b)*E+e] = emb[idx[b*S+s]*E+e]
// ---------------------------------------------------------------------------
__global__ void gather_emb_kernel(const int* __restrict__ idx,
                                  const float* __restrict__ emb,
                                  unsigned short* __restrict__ out, int S) {
  size_t i = (size_t)blockIdx.x * blockDim.x + threadIdx.x;
  size_t total = (size_t)S * WS_B * WS_EMB;
  if (i >= total) return;
  int e = (int)(i & (WS_EMB - 1));
  int row = (int)(i >> 9);
  int b = row & (WS_B - 1);
  int s = row >> 5;
  out[i] = f2bfu(emb[(size_t)idx[b * S + s] * WS_EMB + e]);
}

// ---------------------------------------------------------------------------
// Generic WMMA GEMM: C[M,N] = A[M,K] * W[N,K]^T (+bias1 +bias2) (opt tanh)
// A, W bf16; C fp32. Block = 8 waves arranged 4(M) x 2(N); each wave computes
// a 16x64 strip (4 accumulators, A fragment reused 4x). Block tile 64x128.
// Two-stage software pipeline: loads for k-step s+1 issued before the WMMAs
// of k-step s, so s_wait_loadcnt covers latency with 4 in-flight WMMAs.
// Requires N%128==0, K%64==0 (true at all call sites); M guarded.
// ---------------------------------------------------------------------------
__global__ __launch_bounds__(256)
void gemm_wmma_kernel(const unsigned short* __restrict__ A, int lda,
                      const unsigned short* __restrict__ W, int ldw,
                      float* __restrict__ C, int ldc,
                      const float* __restrict__ bias1,
                      const float* __restrict__ bias2, int M, int K, int act) {
  const int lane = threadIdx.x & 31;
  const int wave = threadIdx.x >> 5;
  const int l15 = lane & 15, lhi = lane >> 4;
  const int m0 = blockIdx.y * 64 + (wave >> 1) * 16;
  const int n0 = blockIdx.x * 128 + (wave & 1) * 64;
  const int mA = (m0 + l15 < M) ? (m0 + l15) : (M - 1);
  const unsigned short* ap = A + (size_t)mA * lda + lhi * 16;
  const unsigned short* wp0 = W + (size_t)(n0 + l15) * ldw + lhi * 16;
  const unsigned short* wp1 = wp0 + (size_t)16 * ldw;
  const unsigned short* wp2 = wp0 + (size_t)32 * ldw;
  const unsigned short* wp3 = wp0 + (size_t)48 * ldw;
  v8f acc0 = vzero8(), acc1 = vzero8(), acc2 = vzero8(), acc3 = vzero8();

  // pipeline prologue: stage X <- k-step 0
  v16bf aX = load16bf(ap);
  v16bf bX0 = load16bf(wp0), bX1 = load16bf(wp1);
  v16bf bX2 = load16bf(wp2), bX3 = load16bf(wp3);
  v16bf aY, bY0, bY1, bY2, bY3;
  const int steps = K >> 5;  // even at every call site
  int kk = 32;
  for (int s = 0; s + 2 < steps; s += 2, kk += 64) {
    // load stage Y (k-step s+1) while WMMAs consume stage X (k-step s)
    aY = load16bf(ap + kk);
    bY0 = load16bf(wp0 + kk);
    bY1 = load16bf(wp1 + kk);
    bY2 = load16bf(wp2 + kk);
    bY3 = load16bf(wp3 + kk);
    __builtin_prefetch(wp0 + kk + 192, 0, 0);
    __builtin_prefetch(wp1 + kk + 192, 0, 0);
    __builtin_prefetch(wp2 + kk + 192, 0, 0);
    __builtin_prefetch(wp3 + kk + 192, 0, 0);
    acc0 = wmma_bf16(aX, bX0, acc0);
    acc1 = wmma_bf16(aX, bX1, acc1);
    acc2 = wmma_bf16(aX, bX2, acc2);
    acc3 = wmma_bf16(aX, bX3, acc3);
    // load stage X (k-step s+2) while WMMAs consume stage Y (k-step s+1)
    aX = load16bf(ap + kk + 32);
    bX0 = load16bf(wp0 + kk + 32);
    bX1 = load16bf(wp1 + kk + 32);
    bX2 = load16bf(wp2 + kk + 32);
    bX3 = load16bf(wp3 + kk + 32);
    acc0 = wmma_bf16(aY, bY0, acc0);
    acc1 = wmma_bf16(aY, bY1, acc1);
    acc2 = wmma_bf16(aY, bY2, acc2);
    acc3 = wmma_bf16(aY, bY3, acc3);
  }
  // epilogue: last two k-steps (kk == (steps-1)*32 here)
  aY = load16bf(ap + kk);
  bY0 = load16bf(wp0 + kk);
  bY1 = load16bf(wp1 + kk);
  bY2 = load16bf(wp2 + kk);
  bY3 = load16bf(wp3 + kk);
  acc0 = wmma_bf16(aX, bX0, acc0);
  acc1 = wmma_bf16(aX, bX1, acc1);
  acc2 = wmma_bf16(aX, bX2, acc2);
  acc3 = wmma_bf16(aX, bX3, acc3);
  acc0 = wmma_bf16(aY, bY0, acc0);
  acc1 = wmma_bf16(aY, bY1, acc1);
  acc2 = wmma_bf16(aY, bY2, acc2);
  acc3 = wmma_bf16(aY, bY3, acc3);

  const int mrow = m0 + lhi * 8;
#pragma unroll
  for (int jn = 0; jn < 4; jn++) {
    v8f av = (jn == 0) ? acc0 : (jn == 1) ? acc1 : (jn == 2) ? acc2 : acc3;
    const int n = n0 + jn * 16 + l15;
    float bs = 0.f;
    if (bias1) bs += bias1[n];
    if (bias2) bs += bias2[n];
#pragma unroll
    for (int vg = 0; vg < 8; vg++) {
      float v = av[vg] + bs;
      if (act == 1) v = tanhf_(v);
      if (mrow + vg < M) C[(size_t)(mrow + vg) * ldc + n] = v;
    }
  }
}

// ---------------------------------------------------------------------------
// Encoder LSTM recurrence (one direction per block; grid = {fwd,bwd}).
// g[t] = gpre[t] + h @ Whh^T ;  i,f,g,o LSTM cell.
// 16 waves: wave = (r: batch row tile) x (hg: 4 hidden tiles).
// Hidden state LDS-resident bf16; cell state in VGPRs across all 64 steps.
// ---------------------------------------------------------------------------
__global__ __launch_bounds__(512)
void enc_lstm_kernel(const float* __restrict__ gpreF,
                     const float* __restrict__ gpreB,
                     const unsigned short* __restrict__ WhhF,   // bf16 [2048,512]
                     const unsigned short* __restrict__ WhhB,
                     const float* __restrict__ h0base,
                     const float* __restrict__ c0base, int stateIdx0,
                     unsigned short* __restrict__ xout,  // bf16 [S,B,1024], col dir*512
                     float* __restrict__ finalH,         // src_h_t base [4,32,512]
                     float* __restrict__ finalC) {
  __shared__ unsigned short lds_h[WS_B * WS_SH];  // 32KB bf16 hidden
  const int dir = blockIdx.x;
  const bool rev = (dir == 1);
  const float* gpre = dir ? gpreB : gpreF;
  const unsigned short* Whh = dir ? WhhB : WhhF;
  const int sidx = stateIdx0 + dir;
  const int colOff = dir * WS_SH;
  const int lane = threadIdx.x & 31;
  const int wave = threadIdx.x >> 5;
  const int l15 = lane & 15, lhi = lane >> 4;
  const int r = wave & 1, hg = wave >> 1;

  for (int i = threadIdx.x; i < WS_B * WS_SH; i += 512)
    lds_h[i] = f2bfu(h0base[(size_t)sidx * WS_B * WS_SH + i]);

  v8f c[4];
#pragma unroll
  for (int j = 0; j < 4; j++) {
    int ht = hg * 4 + j;
#pragma unroll
    for (int vg = 0; vg < 8; vg++) {
      int b = r * 16 + vg + lhi * 8;
      int d = ht * 16 + l15;
      c[j][vg] = c0base[(size_t)sidx * WS_B * WS_SH + b * WS_SH + d];
    }
  }
  __syncthreads();

  const unsigned short* ap = lds_h + (r * 16 + l15) * WS_SH + lhi * 16;
  for (int it = 0; it < 64; it++) {
    const int t = rev ? (63 - it) : it;
    v8f acc[4][4];
#pragma unroll
    for (int j = 0; j < 4; j++)
#pragma unroll
      for (int g = 0; g < 4; g++) acc[j][g] = vzero8();

    for (int kk = 0; kk < WS_SH; kk += 32) {
      v16bf a = load16bf(ap + kk);
#pragma unroll
      for (int j = 0; j < 4; j++) {
        int ht = hg * 4 + j;
#pragma unroll
        for (int g = 0; g < 4; g++) {
          int n = g * WS_SH + ht * 16 + l15;
          v16bf bf = load16bf(Whh + (size_t)n * WS_SH + kk + lhi * 16);
          acc[j][g] = wmma_bf16(a, bf, acc[j][g]);
        }
      }
    }
    __syncthreads();  // all LDS-h reads done before overwrite

#pragma unroll
    for (int j = 0; j < 4; j++) {
      int ht = hg * 4 + j;
#pragma unroll
      for (int vg = 0; vg < 8; vg++) {
        int b = r * 16 + vg + lhi * 8;
        int d = ht * 16 + l15;
        const float* gp = gpre + (size_t)(t * WS_B + b) * (4 * WS_SH);
        float gi = acc[j][0][vg] + gp[d];
        float gf = acc[j][1][vg] + gp[WS_SH + d];
        float gg = acc[j][2][vg] + gp[2 * WS_SH + d];
        float go = acc[j][3][vg] + gp[3 * WS_SH + d];
        float cv = sigmoidf_(gf) * c[j][vg] + sigmoidf_(gi) * tanhf_(gg);
        c[j][vg] = cv;
        float hv = sigmoidf_(go) * tanhf_(cv);
        unsigned short hb = f2bfu(hv);
        lds_h[b * WS_SH + d] = hb;
        xout[(size_t)(t * WS_B + b) * WS_HID + colOff + d] = hb;
        if (it == 63) {
          finalH[(size_t)sidx * WS_B * WS_SH + b * WS_SH + d] = hv;
          finalC[(size_t)sidx * WS_B * WS_SH + b * WS_SH + d] = cv;
        }
      }
    }
    __syncthreads();
  }
}

// ---------------------------------------------------------------------------
// ht = cat(hb1, hf1) (bf16, GEMM A-operand); ct = cat(cb1, cf1) (fp32)
// ---------------------------------------------------------------------------
__global__ void build_htct_kernel(const float* __restrict__ sht,
                                  const float* __restrict__ sct,
                                  unsigned short* __restrict__ ht,
                                  float* __restrict__ ct) {
  int i = blockIdx.x * blockDim.x + threadIdx.x;
  if (i >= WS_B * WS_HID) return;
  int b = i >> 10, k = i & (WS_HID - 1);
  int src = (k < WS_SH) ? (3 * WS_B * WS_SH + b * WS_SH + k)
                        : (2 * WS_B * WS_SH + b * WS_SH + (k - WS_SH));
  ht[i] = f2bfu(sht[src]);
  ct[i] = sct[src];
}

// ---------------------------------------------------------------------------
// Decoder: 64 sequential steps of LSTM cell + soft-dot attention.
// Single persistent workgroup, 32 waves. LDS: h_tilde / hy / (target|wc) bf16
// + scores fp32 (~200KB of the 320KB WGP LDS). c carried in VGPRs.
// ---------------------------------------------------------------------------
__global__ __launch_bounds__(1024)
void decoder_kernel(const float* __restrict__ gpreD,              // [T,B,4096]
                    const unsigned short* __restrict__ Whhd,      // bf16 [4096,1024]
                    const unsigned short* __restrict__ ctx,       // bf16 [S,B,1024]
                    const unsigned short* __restrict__ Wlin_in,   // bf16 [1024,1024]
                    const unsigned short* __restrict__ Wlin_out,  // bf16 [1024,2048]
                    const float* __restrict__ dech0,              // fp32 [32,1024]
                    const float* __restrict__ ct0,                // fp32 [32,1024]
                    unsigned short* __restrict__ trgh) {          // bf16 [B,T,1024]
  __shared__ unsigned short lds_h[WS_B * WS_HID];   // 64KB  recurrent h_tilde
  __shared__ unsigned short lds_hy[WS_B * WS_HID];  // 64KB  cell output hy
  __shared__ unsigned short lds_tw[WS_B * WS_HID];  // 64KB  target, then wc
  __shared__ float lds_sc[WS_B * WS_S];             // 8KB   scores / attn
  const int tid = threadIdx.x;
  const int lane = tid & 31, wave = tid >> 5;
  const int l15 = lane & 15, lhi = lane >> 4;
  const int r = wave & 1, hg = wave >> 1;  // hg in 0..15

  for (int i = tid; i < WS_B * WS_HID; i += 1024) lds_h[i] = f2bfu(dech0[i]);
  v8f c[4];
#pragma unroll
  for (int j = 0; j < 4; j++)
#pragma unroll
    for (int vg = 0; vg < 8; vg++) {
      int b = r * 16 + vg + lhi * 8;
      int d = (hg * 4 + j) * 16 + l15;
      c[j][vg] = ct0[b * WS_HID + d];
    }
  __syncthreads();

  for (int t = 0; t < 64; t++) {
    // ---- Phase A: gates = gpre[t] + h_tilde @ Whhd^T ;  LSTM cell -> hy ----
    {
      v8f acc[4][4];
#pragma unroll
      for (int j = 0; j < 4; j++)
#pragma unroll
        for (int g = 0; g < 4; g++) acc[j][g] = vzero8();
      const unsigned short* ap = lds_h + (r * 16 + l15) * WS_HID + lhi * 16;
      for (int kk = 0; kk < WS_HID; kk += 32) {
        v16bf a = load16bf(ap + kk);
#pragma unroll
        for (int j = 0; j < 4; j++) {
          int ht = hg * 4 + j;
#pragma unroll
          for (int g = 0; g < 4; g++) {
            int n = g * WS_HID + ht * 16 + l15;
            v16bf bf = load16bf(Whhd + (size_t)n * WS_HID + kk + lhi * 16);
            acc[j][g] = wmma_bf16(a, bf, acc[j][g]);
          }
        }
      }
      __syncthreads();
#pragma unroll
      for (int j = 0; j < 4; j++) {
        int ht = hg * 4 + j;
#pragma unroll
        for (int vg = 0; vg < 8; vg++) {
          int b = r * 16 + vg + lhi * 8;
          int d = ht * 16 + l15;
          const float* gp = gpreD + (size_t)(t * WS_B + b) * (4 * WS_HID);
          float gi = acc[j][0][vg] + gp[d];
          float gf = acc[j][1][vg] + gp[WS_HID + d];
          float gg = acc[j][2][vg] + gp[2 * WS_HID + d];
          float go = acc[j][3][vg] + gp[3 * WS_HID + d];
          float cv = sigmoidf_(gf) * c[j][vg] + sigmoidf_(gi) * tanhf_(gg);
          c[j][vg] = cv;
          lds_hy[b * WS_HID + d] = f2bfu(sigmoidf_(go) * tanhf_(cv));
        }
      }
      __syncthreads();
    }
    // ---- Phase B: target = hy @ Wlin_in^T  -> lds_tw ----
    {
      v8f tac[4];
#pragma unroll
      for (int j = 0; j < 4; j++) tac[j] = vzero8();
      const unsigned short* ap = lds_hy + (r * 16 + l15) * WS_HID + lhi * 16;
      for (int kk = 0; kk < WS_HID; kk += 32) {
        v16bf a = load16bf(ap + kk);
#pragma unroll
        for (int j = 0; j < 4; j++) {
          int n = (hg * 4 + j) * 16 + l15;
          v16bf bf = load16bf(Wlin_in + (size_t)n * WS_HID + kk + lhi * 16);
          tac[j] = wmma_bf16(a, bf, tac[j]);
        }
      }
#pragma unroll
      for (int j = 0; j < 4; j++)
#pragma unroll
        for (int vg = 0; vg < 8; vg++) {
          int b = r * 16 + vg + lhi * 8;
          int d = (hg * 4 + j) * 16 + l15;
          lds_tw[b * WS_HID + d] = f2bfu(tac[j][vg]);
        }
      __syncthreads();
    }
    // ---- Phase C: scores[b,s] = ctx[s,b,:] . target[b,:]  (packed bf16) ----
    for (int o = tid; o < WS_B * WS_S; o += 1024) {
      int b = o >> 6, s = o & 63;
      const unsigned int* cp =
          (const unsigned int*)(ctx + (size_t)(s * WS_B + b) * WS_HID);
      const unsigned int* tp = (const unsigned int*)(lds_tw + b * WS_HID);
      float dot = 0.f;
      for (int d = 0; d < WS_HID / 2; d++) {
        unsigned int cu = cp[d], tu = tp[d];
        dot += bflo(cu) * bflo(tu) + bfhi(cu) * bfhi(tu);
      }
      lds_sc[b * WS_S + s] = dot;
    }
    __syncthreads();
    // ---- Phase D: softmax over s, one wave per batch ----
    {
      int b = wave;
      float x0 = lds_sc[b * WS_S + lane];
      float x1 = lds_sc[b * WS_S + 32 + lane];
      float mx = fmaxf(x0, x1);
      for (int off = 16; off; off >>= 1) mx = fmaxf(mx, __shfl_xor(mx, off, 32));
      float e0 = __expf(x0 - mx), e1 = __expf(x1 - mx);
      float sm = e0 + e1;
      for (int off = 16; off; off >>= 1) sm += __shfl_xor(sm, off, 32);
      lds_sc[b * WS_S + lane] = e0 / sm;
      lds_sc[b * WS_S + 32 + lane] = e1 / sm;
    }
    __syncthreads();
    // ---- Phase E: wc[b,d] = sum_s attn[b,s] * ctx[s,b,d]  -> lds_tw ----
    {
      int d = tid;  // 0..1023
      for (int b = 0; b < WS_B; b++) {
        float acw = 0.f;
        for (int s = 0; s < WS_S; s++)
          acw += lds_sc[b * WS_S + s] *
                 bfu2f(ctx[(size_t)(s * WS_B + b) * WS_HID + d]);
        lds_tw[b * WS_HID + d] = f2bfu(acw);
      }
    }
    __syncthreads();
    // ---- Phase F: h_tilde = tanh(cat(wc, hy) @ Wlin_out^T) ----
    {
      v8f hac[4];
#pragma unroll
      for (int j = 0; j < 4; j++) hac[j] = vzero8();
      for (int kk = 0; kk < 2 * WS_HID; kk += 32) {
        const unsigned short* src =
            (kk < WS_HID) ? (lds_tw + (r * 16 + l15) * WS_HID + kk)
                          : (lds_hy + (r * 16 + l15) * WS_HID + (kk - WS_HID));
        v16bf a = load16bf(src + lhi * 16);
#pragma unroll
        for (int j = 0; j < 4; j++) {
          int n = (hg * 4 + j) * 16 + l15;
          v16bf bf = load16bf(Wlin_out + (size_t)n * (2 * WS_HID) + kk + lhi * 16);
          hac[j] = wmma_bf16(a, bf, hac[j]);
        }
      }
#pragma unroll
      for (int j = 0; j < 4; j++)
#pragma unroll
        for (int vg = 0; vg < 8; vg++) {
          int b = r * 16 + vg + lhi * 8;
          int d = (hg * 4 + j) * 16 + l15;
          float hv = tanhf_(hac[j][vg]);
          unsigned short hb = f2bfu(hv);
          lds_h[b * WS_HID + d] = hb;
          trgh[(size_t)(b * WS_S + t) * WS_HID + d] = hb;
        }
      __syncthreads();
    }
  }
}

// ---------------------------------------------------------------------------
extern "C" void kernel_launch(void* const* d_in, const int* in_sizes, int n_in,
                              void* d_out, int out_size, void* d_ws,
                              size_t ws_size, hipStream_t stream) {
  (void)in_sizes; (void)n_in; (void)out_size; (void)ws_size;
  const int* input_src = (const int*)d_in[0];
  const int* input_trg = (const int*)d_in[1];
  const float* h_enc = (const float*)d_in[2];
  const float* c_enc = (const float*)d_in[3];
  const float* emb = (const float*)d_in[4];
  const float* Wih0f = (const float*)d_in[5];
  const float* Whh0f = (const float*)d_in[6];
  const float* bih0f = (const float*)d_in[7];
  const float* bhh0f = (const float*)d_in[8];
  const float* Wih0b = (const float*)d_in[9];
  const float* Whh0b = (const float*)d_in[10];
  const float* bih0b = (const float*)d_in[11];
  const float* bhh0b = (const float*)d_in[12];
  const float* Wih1f = (const float*)d_in[13];
  const float* Whh1f = (const float*)d_in[14];
  const float* bih1f = (const float*)d_in[15];
  const float* bhh1f = (const float*)d_in[16];
  const float* Wih1b = (const float*)d_in[17];
  const float* Whh1b = (const float*)d_in[18];
  const float* bih1b = (const float*)d_in[19];
  const float* bhh1b = (const float*)d_in[20];
  const float* We2d = (const float*)d_in[21];
  const float* be2d = (const float*)d_in[22];
  const float* Wihd = (const float*)d_in[23];
  const float* Whhd = (const float*)d_in[24];
  const float* bihd = (const float*)d_in[25];
  const float* bhhd = (const float*)d_in[26];
  const float* Wlin_in = (const float*)d_in[27];
  const float* Wlin_out = (const float*)d_in[28];
  const float* Wv = (const float*)d_in[29];
  const float* bv = (const float*)d_in[30];

  float* out = (float*)d_out;
  float* logits = out;                                       // [32,64,32000]
  float* srcht = out + (size_t)WS_B * WS_S * WS_V;           // [4,32,512]
  float* srcct = srcht + 4 * WS_B * WS_SH;                   // [4,32,512]

  // ---- workspace layout: fp32 region, then bf16 region (~158 MB total) ----
  float* wsf = (float*)d_ws;
  float* gbuf = wsf;                 // 8,388,608 f  (gate pre-activations, reused)
  float* ct = gbuf + 8388608;        // 32,768 f
  float* dech0 = ct + 32768;         // 32,768 f
  unsigned short* wsu = (unsigned short*)(dech0 + 32768);
  unsigned short* src_emb = wsu;                 // 1,048,576  [S,B,512]
  unsigned short* trg_emb = src_emb + 1048576;   // 1,048,576  [T,B,512]
  unsigned short* x1 = trg_emb + 1048576;        // 2,097,152  [S,B,1024]
  unsigned short* srch = x1 + 2097152;           // 2,097,152  [S,B,1024] (ctx)
  unsigned short* trgh = srch + 2097152;         // 2,097,152  [B,T,1024]
  unsigned short* htb = trgh + 2097152;          // 32,768
  // bf16 weights
  unsigned short* wih0f_b = htb + 32768;
  unsigned short* whh0f_b = wih0f_b + 1048576;
  unsigned short* wih0b_b = whh0f_b + 1048576;
  unsigned short* whh0b_b = wih0b_b + 1048576;
  unsigned short* wih1f_b = whh0b_b + 1048576;
  unsigned short* whh1f_b = wih1f_b + 2097152;
  unsigned short* wih1b_b = whh1f_b + 1048576;
  unsigned short* whh1b_b = wih1b_b + 2097152;
  unsigned short* we2d_b = whh1b_b + 1048576;
  unsigned short* wihd_b = we2d_b + 1048576;
  unsigned short* whhd_b = wihd_b + 2097152;
  unsigned short* wlin_in_b = whhd_b + 4194304;
  unsigned short* wlin_out_b = wlin_in_b + 1048576;
  unsigned short* wv_b = wlin_out_b + 2097152;   // 32,768,000

  float* g0f = gbuf;
  float* g0b = gbuf + 4194304;

  // 0) weight conversions (one streaming pass each)
  auto cvt = [&](const float* s, unsigned short* d, int n) {
    int blocks = (n + 2047) / 2048;
    if (blocks > 2048) blocks = 2048;
    cvt_bf16_kernel<<<dim3(blocks), dim3(512), 0, stream>>>(s, d, n);
  };
  cvt(Wih0f, wih0f_b, 2048 * 512);
  cvt(Whh0f, whh0f_b, 2048 * 512);
  cvt(Wih0b, wih0b_b, 2048 * 512);
  cvt(Whh0b, whh0b_b, 2048 * 512);
  cvt(Wih1f, wih1f_b, 2048 * 1024);
  cvt(Whh1f, whh1f_b, 2048 * 512);
  cvt(Wih1b, wih1b_b, 2048 * 1024);
  cvt(Whh1b, whh1b_b, 2048 * 512);
  cvt(We2d, we2d_b, 1024 * 1024);
  cvt(Wihd, wihd_b, 4096 * 512);
  cvt(Whhd, whhd_b, 4096 * 1024);
  cvt(Wlin_in, wlin_in_b, 1024 * 1024);
  cvt(Wlin_out, wlin_out_b, 1024 * 2048);
  cvt(Wv, wv_b, 32000 * 1024);

  // 1) embedding gathers (time-major, bf16)
  gather_emb_kernel<<<dim3(4096), dim3(256), 0, stream>>>(input_src, emb,
                                                          src_emb, WS_S);
  gather_emb_kernel<<<dim3(4096), dim3(256), 0, stream>>>(input_trg, emb,
                                                          trg_emb, WS_S);
  // 2) encoder layer0 input gates, both dirs: [2048,512]x[512,2048]
  gemm_wmma_kernel<<<dim3(16, 32), dim3(256), 0, stream>>>(
      src_emb, 512, wih0f_b, 512, g0f, 2048, bih0f, bhh0f, 2048, 512, 0);
  gemm_wmma_kernel<<<dim3(16, 32), dim3(256), 0, stream>>>(
      src_emb, 512, wih0b_b, 512, g0b, 2048, bih0b, bhh0b, 2048, 512, 0);
  // 3) layer0 recurrence -> x1 (finals into src_h_t/src_c_t idx 0,1)
  enc_lstm_kernel<<<dim3(2), dim3(512), 0, stream>>>(
      g0f, g0b, whh0f_b, whh0b_b, h_enc, c_enc, 0, x1, srcht, srcct);
  // 4) layer1 input gates: [2048,1024]x[1024,2048]
  gemm_wmma_kernel<<<dim3(16, 32), dim3(256), 0, stream>>>(
      x1, 1024, wih1f_b, 1024, g0f, 2048, bih1f, bhh1f, 2048, 1024, 0);
  gemm_wmma_kernel<<<dim3(16, 32), dim3(256), 0, stream>>>(
      x1, 1024, wih1b_b, 1024, g0b, 2048, bih1b, bhh1b, 2048, 1024, 0);
  // 5) layer1 recurrence -> srch/ctx (finals idx 2,3)
  enc_lstm_kernel<<<dim3(2), dim3(512), 0, stream>>>(
      g0f, g0b, whh1f_b, whh1b_b, h_enc, c_enc, 2, srch, srcht, srcct);
  // 6) h_t = cat(hb1,hf1) bf16, c_t = cat(cb1,cf1) fp32
  build_htct_kernel<<<dim3(128), dim3(256), 0, stream>>>(srcht, srcct, htb, ct);
  // 7) dec_h0 = tanh(h_t @ We2d^T + be2d)   (M=32, guarded)
  gemm_wmma_kernel<<<dim3(8, 1), dim3(256), 0, stream>>>(
      htb, 1024, we2d_b, 1024, dech0, 1024, be2d, nullptr, 32, 1024, 1);
  // 8) decoder input gates: [2048,512]x[512,4096] -> gbuf (full reuse)
  gemm_wmma_kernel<<<dim3(32, 32), dim3(256), 0, stream>>>(
      trg_emb, 512, wihd_b, 512, gbuf, 4096, bihd, bhhd, 2048, 512, 0);
  // 9) decoder recurrence + attention
  decoder_kernel<<<dim3(1), dim3(1024), 0, stream>>>(
      gbuf, whhd_b, srch, wlin_in_b, wlin_out_b, dech0, ct, trgh);
  // 10) vocab projection: [2048,1024]x[1024,32000] (+bv) -> logits
  gemm_wmma_kernel<<<dim3(250, 32), dim3(256), 0, stream>>>(
      trgh, 1024, wv_b, 1024, logits, 32000, bv, nullptr, 2048, 1024, 0);
}
